// ZAM_WAM_Layer_17446157156898
// MI455X (gfx1250) — matrix operand order repaired and checked by reading, not compiled, not run on hardware
//
#include <hip/hip_runtime.h>
#include <hip/hip_bf16.h>
#include <cstdint>

typedef __attribute__((ext_vector_type(16))) _Float16 v16h;
typedef __attribute__((ext_vector_type(8)))  _Float16 v8h;
typedef __attribute__((ext_vector_type(8)))  float    v8f;

// ---------------- problem constants ----------------
#define BB    8
#define CIN1  256
#define CMID  128
#define HH    128
#define WW    128
#define XT    130      // padded row width (x = -1 .. 128)
#define CP    72       // LDS channel pitch for a 64-channel group (+8 pad halves)
#define CG    64       // channels staged per group
#define CP2   136      // padded channel pitch of f16 NHWC intermediate in workspace

// ---------------- workspace layout (bytes) ----------------
static constexpr size_t WP1_OFF = 0;
static constexpr size_t WP2_OFF = WP1_OFF + (size_t)9*128*256*2;   // 589824
static constexpr size_t S1_OFF  = WP2_OFF + (size_t)9*128*128*2;   // 884736
static constexpr size_t B1_OFF  = S1_OFF + 512;
static constexpr size_t S2_OFF  = B1_OFF + 512;
static constexpr size_t B2_OFF  = S2_OFF + 512;
static constexpr size_t ZS_OFF  = B2_OFF + 512;
static constexpr size_t HB_OFF  = 917504;                          // 896 KB, 16B aligned

// ---------------- prep: pack weights to f16, fold BN ----------------
__global__ __launch_bounds__(256) void zam_prep_pack(
    const float* __restrict__ w1, const float* __restrict__ w2,
    const float* __restrict__ cb1, const float* __restrict__ bw1,
    const float* __restrict__ bb1, const float* __restrict__ bm1, const float* __restrict__ bv1,
    const float* __restrict__ cb2, const float* __restrict__ bw2,
    const float* __restrict__ bb2, const float* __restrict__ bm2, const float* __restrict__ bv2,
    char* __restrict__ ws) {
  _Float16* wp1 = (_Float16*)(ws + WP1_OFF);
  _Float16* wp2 = (_Float16*)(ws + WP2_OFF);
  float* s1 = (float*)(ws + S1_OFF); float* b1 = (float*)(ws + B1_OFF);
  float* s2 = (float*)(ws + S2_OFF); float* b2 = (float*)(ws + B2_OFF);
  int idx = blockIdx.x * 256 + threadIdx.x;
  const int N1 = 9*128*256, N2 = 9*128*128;
  if (idx < N1) {
    int tp = idx / (128*256); int n = (idx / 256) & 127; int k = idx & 255;
    wp1[idx] = (_Float16)w1[((size_t)n*256 + k)*9 + tp];       // OIHW -> [tap][o][i]
  } else if (idx < N1 + N2) {
    int j = idx - N1;
    int tp = j / (128*128); int n = (j / 128) & 127; int k = j & 127;
    wp2[j] = (_Float16)w2[((size_t)n*128 + k)*9 + tp];
  }
  if (blockIdx.x == 0 && threadIdx.x < 128) {
    int c = threadIdx.x;
    float sc1 = bw1[c] * rsqrtf(bv1[c] + 1e-5f);
    s1[c] = sc1; b1[c] = (cb1[c] - bm1[c]) * sc1 + bb1[c];
    float sc2 = bw2[c] * rsqrtf(bv2[c] + 1e-5f);
    s2[c] = sc2; b2[c] = (cb2[c] - bm2[c]) * sc2 + bb2[c];
  }
}

// ---------------- prep: Zernike constant ----------------
// c = (p+1)/(pi*(N-1)^2) * sum_{i,j} cos(q*atan2(j,i)),  p=2, q=1, N=128
__global__ __launch_bounds__(256) void zam_prep_zern(const float* __restrict__ gamma,
                                                     char* __restrict__ ws) {
  __shared__ float red[256];
  int t = threadIdx.x;
  float s = 0.f;
  for (int e = t; e < 128*128; e += 256) {
    int i = e >> 7, j = e & 127;
    s += (i == 0 && j == 0) ? 1.0f : (float)i * rsqrtf((float)(i*i + j*j));
  }
  red[t] = s; __syncthreads();
  for (int off = 128; off > 0; off >>= 1) {
    if (t < off) red[t] += red[t + off];
    __syncthreads();
  }
  if (t == 0) {
    float c = red[0] * 3.0f / (3.14159265358979f * 127.0f * 127.0f);
    *(float*)(ws + ZS_OFF) = 1.0f + gamma[0] * c;
  }
}

// ---------------- conv1 (256->128) + BN + PReLU + ZAM, h -> f16 NHWC ----------------
__global__ __launch_bounds__(256) void zam_conv1(const float* __restrict__ xin,
                                                 const float* __restrict__ a1p,
                                                 char* __restrict__ ws) {
  __shared__ __align__(16) _Float16 lin[3 * XT * CP];   // 56,160 B
  const _Float16* wp1 = (const _Float16*)(ws + WP1_OFF);
  const float* s1 = (const float*)(ws + S1_OFF);
  const float* b1 = (const float*)(ws + B1_OFF);
  const float zscale = *(const float*)(ws + ZS_OFF);
  const float a1 = a1p[0];

  const int blk = blockIdx.x;            // b*128 + y
  const int b = blk >> 7, y = blk & 127;
  const int t0 = threadIdx.x;
  const int lane = t0 & 31, wid = t0 >> 5;
  const int hf = lane >> 4, nl = lane & 15;
  const int wm = wid * 16;

  v8f acc[8] = {};

  for (int g = 0; g < 4; ++g) {
    __syncthreads();
    // zero-fill tile (covers halo + OOB rows)
    unsigned int* lz = (unsigned int*)lin;
    for (int e = t0; e < 3 * XT * CP / 2; e += 256) lz[e] = 0u;
    __syncthreads();
    // stage 3 input rows, 64 channels, f32 -> f16, [x][c] layout
    for (int r = 0; r < 3; ++r) {
      int yy = y + r - 1;
      if (yy < 0 || yy >= HH) continue;
      const float* src = xin + (((size_t)b * CIN1 + (size_t)g * CG) * HH + yy) * WW;
      for (int e = t0; e < CG * WW; e += 256) {
        int xx = e & 127, cl = e >> 7;
        lin[(r * XT + xx + 1) * CP + cl] = (_Float16)src[(size_t)cl * HH * WW + xx];
      }
    }
    __syncthreads();
    // implicit GEMM: 9 taps x 2 K-chunks of 32
    for (int tp = 0; tp < 9; ++tp) {
      int dy = tp / 3, dx = tp % 3;
      __builtin_prefetch(&wp1[(size_t)(tp * CMID) * CIN1 + g * CG], 0, 1);
      #pragma unroll
      for (int kc = 0; kc < 2; ++kc) {
        int k0 = kc * 32;
        int xi = wm + nl + dx;                       // output x + (dx-1) + 1
        int ab = (dy * XT + xi) * CP + k0 + hf * 16; // A: 16 contiguous K per lane half
        union { v16h v; v8h h[2]; } ua;
        ua.h[0] = *(const v8h*)&lin[ab];
        ua.h[1] = *(const v8h*)&lin[ab + 8];
        int kabs = g * CG + k0 + hf * 16;            // B: 16 contiguous K per lane half, same order as A
        #pragma unroll
        for (int n = 0; n < 8; ++n) {
          const v16h bf = *(const v16h*)&wp1[((size_t)(tp * CMID + n * 16 + nl)) * CIN1 + kabs];
          acc[n] = __builtin_amdgcn_wmma_f32_16x16x32_f16(
              false, ua.v, false, bf, (short)0, acc[n], false, false);
        }
      }
    }
  }

  // epilogue: BN + PReLU + ZAM scale, stage f16 through LDS
  __syncthreads();
  _Float16* lout = lin;                              // reuse as [x][CP2]
  #pragma unroll
  for (int n = 0; n < 8; ++n) {
    int c = n * 16 + nl;
    float sc = s1[c], bs = b1[c];
    #pragma unroll
    for (int i = 0; i < 8; ++i) {
      float v = acc[n][i] * sc + bs;
      v = v > 0.f ? v : a1 * v;
      v *= zscale;
      lout[(wm + hf * 8 + i) * CP2 + c] = (_Float16)v;
    }
  }
  __syncthreads();
  // async LDS -> global copy of the row tile (bypasses VGPRs, ASYNCcnt-tracked)
  uint4* hb = (uint4*)(ws + HB_OFF);
  size_t base = (size_t)blk * (WW * CP2 / 8);
  for (int e = t0; e < WW * CP2 / 8; e += 256) {
    const uint4* gdst = hb + base + e;
    unsigned lsrc = (unsigned)(uintptr_t)&lin[e * 8];
    asm volatile("global_store_async_from_lds_b128 %0, %1, off"
                 :: "v"(gdst), "v"(lsrc) : "memory");
  }
  asm volatile("s_wait_asynccnt 0" ::: "memory");
}

// ---------------- conv2 (128->128) + BN + PReLU, out f32 NCHW ----------------
__global__ __launch_bounds__(256) void zam_conv2(const float* __restrict__ a2p,
                                                 char* __restrict__ ws,
                                                 float* __restrict__ out) {
  __shared__ __align__(16) _Float16 lin[3 * XT * CP];
  const _Float16* wp2 = (const _Float16*)(ws + WP2_OFF);
  const float* s2 = (const float*)(ws + S2_OFF);
  const float* b2 = (const float*)(ws + B2_OFF);
  const float a2 = a2p[0];

  const int blk = blockIdx.x;
  const int b = blk >> 7, y = blk & 127;
  const int t0 = threadIdx.x;
  const int lane = t0 & 31, wid = t0 >> 5;
  const int hf = lane >> 4, nl = lane & 15;
  const int wm = wid * 16;

  v8f acc[8] = {};
  const uint4* hb = (const uint4*)(ws + HB_OFF);

  for (int g = 0; g < 2; ++g) {
    __syncthreads();
    unsigned int* lz = (unsigned int*)lin;
    for (int e = t0; e < 3 * XT * CP / 2; e += 256) lz[e] = 0u;
    __syncthreads();
    // async global -> LDS staging of the 3 input rows (64 f16 channels / x)
    for (int r = 0; r < 3; ++r) {
      int yy = y + r - 1;
      if (yy < 0 || yy >= HH) continue;
      size_t sb = (size_t)(b * HH + yy) * (WW * CP2 / 8);
      for (int e = t0; e < WW * 8; e += 256) {
        int xx = e >> 3, q = e & 7;                  // 8 x uint4 = 64 f16 channels
        const uint4* gsrc = hb + sb + (size_t)xx * 17 + g * 8 + q;
        unsigned ldst = (unsigned)(uintptr_t)&lin[(r * XT + xx + 1) * CP + q * 8];
        asm volatile("global_load_async_to_lds_b128 %0, %1, off"
                     :: "v"(ldst), "v"(gsrc) : "memory");
      }
    }
    asm volatile("s_wait_asynccnt 0" ::: "memory");
    __syncthreads();
    for (int tp = 0; tp < 9; ++tp) {
      int dy = tp / 3, dx = tp % 3;
      __builtin_prefetch(&wp2[(size_t)(tp * CMID) * CMID + g * CG], 0, 1);
      #pragma unroll
      for (int kc = 0; kc < 2; ++kc) {
        int k0 = kc * 32;
        int xi = wm + nl + dx;
        int ab = (dy * XT + xi) * CP + k0 + hf * 16;
        union { v16h v; v8h h[2]; } ua;
        ua.h[0] = *(const v8h*)&lin[ab];
        ua.h[1] = *(const v8h*)&lin[ab + 8];
        int kabs = g * CG + k0 + hf * 16;
        #pragma unroll
        for (int n = 0; n < 8; ++n) {
          const v16h bf = *(const v16h*)&wp2[((size_t)(tp * CMID + n * 16 + nl)) * CMID + kabs];
          acc[n] = __builtin_amdgcn_wmma_f32_16x16x32_f16(
              false, ua.v, false, bf, (short)0, acc[n], false, false);
        }
      }
    }
  }

  // epilogue in two 64-channel halves ([c][x] f32 staging fits in LDS)
  float* lof = (float*)lin;
  for (int ch = 0; ch < 2; ++ch) {
    __syncthreads();
    #pragma unroll
    for (int n = ch * 4; n < ch * 4 + 4; ++n) {
      int c = n * 16 + nl;
      float sc = s2[c], bs = b2[c];
      #pragma unroll
      for (int i = 0; i < 8; ++i) {
        float v = acc[n][i] * sc + bs;
        v = v > 0.f ? v : a2 * v;
        lof[(c - ch * 64) * WW + wm + hf * 8 + i] = v;
      }
    }
    __syncthreads();
    float4* o4 = (float4*)out;
    const float4* l4 = (const float4*)lof;
    for (int e = t0; e < 64 * WW / 4; e += 256) {
      int cc = e >> 5, xw = e & 31;
      o4[((size_t)(b * CMID + ch * 64 + cc) * HH + y) * (WW / 4) + xw] = l4[e];
    }
  }
}

extern "C" void kernel_launch(void* const* d_in, const int* in_sizes, int n_in,
                              void* d_out, int out_size, void* d_ws, size_t ws_size,
                              hipStream_t stream) {
  const float* x   = (const float*)d_in[0];
  const float* c1w = (const float*)d_in[1];
  const float* c1b = (const float*)d_in[2];
  const float* b1w = (const float*)d_in[3];
  const float* b1b = (const float*)d_in[4];
  const float* b1m = (const float*)d_in[5];
  const float* b1v = (const float*)d_in[6];
  const float* p1a = (const float*)d_in[7];
  const float* gam = (const float*)d_in[8];
  const float* c2w = (const float*)d_in[9];
  const float* c2b = (const float*)d_in[10];
  const float* b2w = (const float*)d_in[11];
  const float* b2b = (const float*)d_in[12];
  const float* b2m = (const float*)d_in[13];
  const float* b2v = (const float*)d_in[14];
  const float* p2a = (const float*)d_in[15];
  char* ws = (char*)d_ws;
  float* out = (float*)d_out;

  zam_prep_pack<<<1728, 256, 0, stream>>>(c1w, c2w, c1b, b1w, b1b, b1m, b1v,
                                          c2b, b2w, b2b, b2m, b2v, ws);
  zam_prep_zern<<<1, 256, 0, stream>>>(gam, ws);
  zam_conv1<<<BB * HH, 256, 0, stream>>>(x, p1a, ws);
  zam_conv2<<<BB * HH, 256, 0, stream>>>(p2a, ws, out);
}
